// Transformer_5454608466561
// MI455X (gfx1250) — matrix operand order repaired and checked by reading, not compiled, not run on hardware
//
#include <hip/hip_runtime.h>
#include <math.h>

#define B_    4
#define L_    1024
#define D_    512
#define H_    8
#define DH_   64
#define DFF_  2048
#define BL_   (B_*L_)
#define NSL_  (B_*H_)

typedef __bf16 bf16;
typedef __attribute__((ext_vector_type(16))) __bf16 v16bf;
typedef __attribute__((ext_vector_type(8)))  __bf16 v8bf;
typedef __attribute__((ext_vector_type(8)))  float  v8f;

#define GEMM_ATTR __attribute__((amdgpu_flat_work_group_size(32, 32), amdgpu_waves_per_eu(1)))

// ---------------------------------------------------------------------------
// WMMA fragment loaders (layouts per CDNA5 ISA 7.12.2, wave32)
// A (16x32 bf16, row-major source): lane l -> row m=l&15;
//   lanes<16: K in {0..7} u {16..23}; lanes>=16: K in {8..15} u {24..31}
// B (32x16 bf16): lane l -> col n=l&15; K same split as A (Bt rows are K-contiguous)
// C/D (16x16 f32): vgpr r -> row r (+8 for lanes>=16), col = lane&15
// ---------------------------------------------------------------------------
__device__ __forceinline__ v16bf load_a_frag(const bf16* __restrict__ A, int lda,
                                             int m, int kk, int lane) {
  const int base0 = (lane & 16) ? 8 : 0;
  const bf16* p = A + (size_t)m * lda + kk + base0;
  v8bf lo = *(const v8bf*)(p);
  v8bf hi = *(const v8bf*)(p + 16);
  return __builtin_shufflevector(lo, hi, 0,1,2,3,4,5,6,7,8,9,10,11,12,13,14,15);
}

__device__ __forceinline__ v16bf load_b_frag(const bf16* __restrict__ Bt, int ldb,
                                             int n, int kk, int lane) {
  const int kbase = (lane & 16) ? 8 : 0;
  const bf16* p = Bt + (size_t)n * ldb + kk + kbase;
  v8bf lo = *(const v8bf*)(p);
  v8bf hi = *(const v8bf*)(p + 16);
  return __builtin_shufflevector(lo, hi, 0,1,2,3,4,5,6,7,8,9,10,11,12,13,14,15);
}

// 64x32 tile per wave: 4 A-frags x 2 B-frags -> 8 WMMAs per K=32 step
__device__ __forceinline__ void wave_gemm_64x32(const bf16* __restrict__ A, int lda,
                                                const bf16* __restrict__ Bt, int ldb,
                                                int Kdim, int m0, int n0, int lane,
                                                v8f acc[4][2]) {
  const int rr = lane & 15;
  for (int kk = 0; kk < Kdim; kk += 32) {
    v16bf a[4], b[2];
#pragma unroll
    for (int i = 0; i < 4; i++) a[i] = load_a_frag(A, lda, m0 + i * 16 + rr, kk, lane);
#pragma unroll
    for (int i = 0; i < 2; i++) b[i] = load_b_frag(Bt, ldb, n0 + i * 16 + rr, kk, lane);
#pragma unroll
    for (int mi = 0; mi < 4; mi++)
#pragma unroll
      for (int ni = 0; ni < 2; ni++)
        acc[mi][ni] = __builtin_amdgcn_wmma_f32_16x16x32_bf16(
            false, a[mi], false, b[ni], (short)0, acc[mi][ni], false, false);
  }
}

__device__ __forceinline__ void zero_acc(v8f acc[4][2]) {
#pragma unroll
  for (int i = 0; i < 4; i++)
#pragma unroll
    for (int j = 0; j < 2; j++)
#pragma unroll
      for (int r = 0; r < 8; r++) acc[i][j][r] = 0.f;
}

// ---------------------------------------------------------------------------
// Generic GEMM: C[M,N] = A_bf16[M,K] @ Bt_bf16[N,K]^T + bias ; optional relu,
// optional f32 and/or bf16 stores. One wave per 64x32 tile.
// ---------------------------------------------------------------------------
__global__ void GEMM_ATTR gemm_kernel(const bf16* __restrict__ A, int lda,
                                      const bf16* __restrict__ Bt, int ldb,
                                      const float* __restrict__ bias,
                                      float* __restrict__ Cf, bf16* __restrict__ Cb,
                                      int ldc, int Kdim, int relu) {
  const int lane = threadIdx.x & 31;
  const int n0 = blockIdx.x * 32;
  const int m0 = blockIdx.y * 64;
  v8f acc[4][2];
  zero_acc(acc);
  wave_gemm_64x32(A, lda, Bt, ldb, Kdim, m0, n0, lane, acc);
  const int roff = (lane & 16) ? 8 : 0;
  const int nc = lane & 15;
#pragma unroll
  for (int mi = 0; mi < 4; mi++)
#pragma unroll
    for (int ni = 0; ni < 2; ni++)
#pragma unroll
      for (int r = 0; r < 8; r++) {
        int m = m0 + mi * 16 + roff + r;
        int n = n0 + ni * 16 + nc;
        float v = acc[mi][ni][r] + bias[n];
        if (relu) v = fmaxf(v, 0.f);
        if (Cf) Cf[(size_t)m * ldc + n] = v;
        if (Cb) Cb[(size_t)m * ldc + n] = (bf16)v;
      }
}

// V projection: C = kv @ Wv + bv, scattered as Vt[b,h,d,j]  (per-head transpose)
__global__ void GEMM_ATTR gemm_vt_kernel(const bf16* __restrict__ A, int lda,
                                         const bf16* __restrict__ Bt, int ldb,
                                         const float* __restrict__ bias,
                                         bf16* __restrict__ Vt, int Kdim) {
  const int lane = threadIdx.x & 31;
  const int n0 = blockIdx.x * 32;
  const int m0 = blockIdx.y * 64;
  v8f acc[4][2];
  zero_acc(acc);
  wave_gemm_64x32(A, lda, Bt, ldb, Kdim, m0, n0, lane, acc);
  const int roff = (lane & 16) ? 8 : 0;
  const int nc = lane & 15;
#pragma unroll
  for (int mi = 0; mi < 4; mi++)
#pragma unroll
    for (int ni = 0; ni < 2; ni++)
#pragma unroll
      for (int r = 0; r < 8; r++) {
        int m = m0 + mi * 16 + roff + r;     // b*L + j
        int n = n0 + ni * 16 + nc;           // h*DH + d
        int bb = m >> 10, j = m & (L_ - 1);
        int h = n >> 6, d = n & (DH_ - 1);
        float v = acc[mi][ni][r] + bias[n];
        Vt[(((size_t)(bb * H_ + h)) * DH_ + d) * L_ + j] = (bf16)v;
      }
}

// logits: S[b,h,q,k] = (Q.K^T * ca + cg*table + mask*NEG) / sqrt(DH)
__global__ void GEMM_ATTR qk_kernel(const bf16* __restrict__ Qbf, const bf16* __restrict__ Kbf,
                                    const float* __restrict__ hb, const float* __restrict__ pi,
                                    const float* __restrict__ mask,
                                    const float* __restrict__ ca_p, const float* __restrict__ cg_p,
                                    float* __restrict__ S) {
  const int lane = threadIdx.x & 31;
  const int slice = blockIdx.z;
  const int b = slice >> 3, h = slice & 7;
  const bf16* A  = Qbf + ((size_t)b * L_) * D_ + h * DH_;
  const bf16* Bt = Kbf + ((size_t)b * L_) * D_ + h * DH_;
  const int n0 = blockIdx.x * 32, m0 = blockIdx.y * 64;
  v8f acc[4][2];
  zero_acc(acc);
  wave_gemm_64x32(A, D_, Bt, D_, DH_, m0, n0, lane, acc);
  const float ca = ca_p[0], cg = cg_p[0];
  const float* tb = (h < (H_ / 2) ? hb : pi) + (size_t)b * L_ * L_;
  const float* mk = mask + (size_t)b * L_ * L_;
  float* Sp = S + (size_t)slice * L_ * L_;
  const int roff = (lane & 16) ? 8 : 0;
  const int nc = lane & 15;
#pragma unroll
  for (int mi = 0; mi < 4; mi++)
#pragma unroll
    for (int ni = 0; ni < 2; ni++)
#pragma unroll
      for (int r = 0; r < 8; r++) {
        int q = m0 + mi * 16 + roff + r;
        int k = n0 + ni * 16 + nc;
        size_t ix = (size_t)q * L_ + k;
        float s = (acc[mi][ni][r] * ca + cg * tb[ix] + mk[ix] * -1e9f) * 0.125f;
        Sp[ix] = s;
      }
}

// out[b,h,i,d] = scale[b,h] * sum_j At[i,j] * Vt[d,j] -> attnm[b,i, h*DH+d] bf16
__global__ void GEMM_ATTR av_kernel(const bf16* __restrict__ At, const bf16* __restrict__ Vt,
                                    const float* __restrict__ gscale, bf16* __restrict__ attnm) {
  const int lane = threadIdx.x & 31;
  const int slice = blockIdx.z;
  const int b = slice >> 3, h = slice & 7;
  const bf16* A  = At + (size_t)slice * L_ * L_;
  const bf16* Bt = Vt + (size_t)slice * DH_ * L_;
  const int n0 = blockIdx.x * 32, m0 = blockIdx.y * 64;
  v8f acc[4][2];
  zero_acc(acc);
  wave_gemm_64x32(A, L_, Bt, L_, L_, m0, n0, lane, acc);
  const float c = gscale[slice];
  const int roff = (lane & 16) ? 8 : 0;
  const int nc = lane & 15;
#pragma unroll
  for (int mi = 0; mi < 4; mi++)
#pragma unroll
    for (int ni = 0; ni < 2; ni++)
#pragma unroll
      for (int r = 0; r < 8; r++) {
        int i = m0 + mi * 16 + roff + r;
        int d = n0 + ni * 16 + nc;
        attnm[((size_t)(b * L_ + i)) * D_ + h * DH_ + d] = (bf16)(acc[mi][ni][r] * c);
      }
}

// ------------------------- softmax reductions ------------------------------
__global__ void pmax_kernel(const float* __restrict__ S, float* __restrict__ pmax) {
  const int slice = blockIdx.y, chunk = blockIdx.x, t = threadIdx.x;
  const float4* p = (const float4*)(S + (size_t)slice * L_ * L_ + (size_t)chunk * 32768);
  float m = -3.4e38f;
  for (int it = 0; it < 32; ++it) {
    float4 v = p[it * 256 + t];
    m = fmaxf(m, fmaxf(fmaxf(v.x, v.y), fmaxf(v.z, v.w)));
  }
  __shared__ float red[256];
  red[t] = m; __syncthreads();
  for (int s2 = 128; s2 > 0; s2 >>= 1) { if (t < s2) red[t] = fmaxf(red[t], red[t + s2]); __syncthreads(); }
  if (t == 0) pmax[slice * 32 + chunk] = red[0];
}

__global__ void fmax_kernel(const float* __restrict__ pmax, float* __restrict__ gmax) {
  const int slice = blockIdx.x, t = threadIdx.x;
  __shared__ float red[32];
  red[t] = pmax[slice * 32 + t]; __syncthreads();
  for (int s2 = 16; s2 > 0; s2 >>= 1) { if (t < s2) red[t] = fmaxf(red[t], red[t + s2]); __syncthreads(); }
  if (t == 0) gmax[slice] = red[0];
}

// exp(S - gmax), transpose [j,i]->[i,j] to bf16 At, partial sums
__global__ void expT_kernel(const float* __restrict__ S, const float* __restrict__ gmax,
                            bf16* __restrict__ At, float* __restrict__ psum) {
  __shared__ float tile[64][65];
  __shared__ float red[256];
  const int slice = blockIdx.z;
  const int i0 = blockIdx.x * 64;   // inner (output-row) index
  const int j0 = blockIdx.y * 64;   // query (S-row) index
  const float gm = gmax[slice];
  const int t = threadIdx.x;
  const int ty = t >> 2;            // local j row 0..63
  const int tx = t & 3;             // 16-col group
  const float* Sp = S + ((size_t)slice * L_ + (j0 + ty)) * L_ + i0 + tx * 16;
  float lsum = 0.f;
#pragma unroll
  for (int g = 0; g < 4; ++g) {
    float4 v = ((const float4*)Sp)[g];
    float e0 = __expf(v.x - gm), e1 = __expf(v.y - gm);
    float e2 = __expf(v.z - gm), e3 = __expf(v.w - gm);
    lsum += (e0 + e1) + (e2 + e3);
    int c = tx * 16 + g * 4;
    tile[c + 0][ty] = e0; tile[c + 1][ty] = e1;
    tile[c + 2][ty] = e2; tile[c + 3][ty] = e3;
  }
  red[t] = lsum; __syncthreads();
  for (int s2 = 128; s2 > 0; s2 >>= 1) { if (t < s2) red[t] += red[t + s2]; __syncthreads(); }
  if (t == 0) psum[(size_t)slice * 256 + blockIdx.y * 16 + blockIdx.x] = red[0];
  // write At[i0+ty][j0 + tx*16 + 0..15]
  bf16* Op = At + ((size_t)slice * L_ + (i0 + ty)) * L_ + j0 + tx * 16;
#pragma unroll
  for (int g = 0; g < 2; ++g) {
    v8bf o;
#pragma unroll
    for (int c2 = 0; c2 < 8; ++c2) o[c2] = (bf16)tile[ty][tx * 16 + g * 8 + c2];
    *((v8bf*)(Op + g * 8)) = o;
  }
}

// final sum + nonzero-count(protok[0,:]) -> gscale = nz / sum
__global__ void fsum_kernel(const float* __restrict__ psum, const int* __restrict__ protok,
                            float* __restrict__ gscale) {
  const int slice = blockIdx.x, t = threadIdx.x;
  __shared__ float red[256];
  __shared__ float rnz[256];
  red[t] = psum[(size_t)slice * 256 + t];
  float nz = 0.f;
  for (int i = t; i < L_; i += 256) nz += (protok[i] != 0) ? 1.f : 0.f;
  rnz[t] = nz; __syncthreads();
  for (int s2 = 128; s2 > 0; s2 >>= 1) {
    if (t < s2) { red[t] += red[t + s2]; rnz[t] += rnz[t + s2]; }
    __syncthreads();
  }
  if (t == 0) gscale[slice] = rnz[0] / red[0];
}

// y = LN(X + Yadd) * g + b  (row = 512 elems, block = 256 threads)
__global__ void add_ln_kernel(const float* __restrict__ X, const float* __restrict__ Yadd,
                              const float* __restrict__ g, const float* __restrict__ bb,
                              float* __restrict__ outF, bf16* __restrict__ outB) {
  const int row = blockIdx.x, t = threadIdx.x;
  const size_t base = (size_t)row * D_;
  float x0 = X[base + t]        + Yadd[base + t];
  float x1 = X[base + t + 256]  + Yadd[base + t + 256];
  __shared__ float red[256];
  red[t] = x0 + x1; __syncthreads();
  for (int s2 = 128; s2 > 0; s2 >>= 1) { if (t < s2) red[t] += red[t + s2]; __syncthreads(); }
  const float mean = red[0] * (1.f / D_);
  __syncthreads();
  float d0 = x0 - mean, d1 = x1 - mean;
  red[t] = d0 * d0 + d1 * d1; __syncthreads();
  for (int s2 = 128; s2 > 0; s2 >>= 1) { if (t < s2) red[t] += red[t + s2]; __syncthreads(); }
  const float rs = rsqrtf(red[0] * (1.f / D_) + 1e-9f);
  float y0 = d0 * rs * g[t]       + bb[t];
  float y1 = d1 * rs * g[t + 256] + bb[t + 256];
  if (outF) { outF[base + t] = y0; outF[base + t + 256] = y1; }
  if (outB) { outB[base + t] = (bf16)y0; outB[base + t + 256] = (bf16)y1; }
}

// f32 -> bf16
__global__ void cvt_kernel(const float* __restrict__ x, bf16* __restrict__ y, int n) {
  int i = blockIdx.x * 256 + threadIdx.x;
  if (i < n) y[i] = (bf16)x[i];
}

// W[K,N] f32 -> Wt[N,K] bf16 (transposed)
__global__ void wconv_kernel(const float* __restrict__ W, bf16* __restrict__ Wt, int K, int N) {
  int idx = blockIdx.x * 256 + threadIdx.x;
  if (idx < K * N) {
    int n = idx / K, k = idx - n * K;
    Wt[idx] = (bf16)W[(size_t)k * N + n];
  }
}

// ---------------------------------------------------------------------------
extern "C" void kernel_launch(void* const* d_in, const int* in_sizes, int n_in,
                              void* d_out, int out_size, void* d_ws, size_t ws_size,
                              hipStream_t stream) {
  (void)in_sizes; (void)n_in; (void)out_size; (void)ws_size;
  const float* pro_inp = (const float*)d_in[0];
  const float* rna_inp = (const float*)d_in[1];
  const float* mask    = (const float*)d_in[2];
  const float* hb      = (const float*)d_in[3];
  const float* pi      = (const float*)d_in[4];
  const int*   protok  = (const int*)d_in[5];
  // params flatten order: wq.w,wq.b, wk.w,wk.b, wv.w,wv.b, wo.w,wo.b,
  // ffn1.w,ffn1.b, ffn2.w,ffn2.b, ln1.g,ln1.b, ln2.g,ln2.b, coeff_att, coeff_aug
  const int PB[2] = {6, 24}; // params_pro, params_rna bases

  size_t off = 0;
  char* wsb = (char*)d_ws;
  auto walloc = [&](size_t bytes) -> void* {
    void* r = wsb + off;
    off = (off + bytes + 255) & ~(size_t)255;
    return r;
  };

  bf16* inpro = (bf16*)walloc((size_t)BL_ * D_ * 2);
  bf16* inrna = (bf16*)walloc((size_t)BL_ * D_ * 2);
  bf16* WqT[2], *WkT[2], *WvT[2], *WoT[2], *W1T[2], *W2T[2];
  for (int s = 0; s < 2; s++) {
    WqT[s] = (bf16*)walloc((size_t)D_ * D_ * 2);
    WkT[s] = (bf16*)walloc((size_t)D_ * D_ * 2);
    WvT[s] = (bf16*)walloc((size_t)D_ * D_ * 2);
    WoT[s] = (bf16*)walloc((size_t)D_ * D_ * 2);
    W1T[s] = (bf16*)walloc((size_t)D_ * DFF_ * 2);
    W2T[s] = (bf16*)walloc((size_t)DFF_ * D_ * 2);
  }
  bf16*  Qbf   = (bf16*)walloc((size_t)BL_ * D_ * 2);
  bf16*  Kbf   = (bf16*)walloc((size_t)BL_ * D_ * 2);
  bf16*  Vt    = (bf16*)walloc((size_t)NSL_ * DH_ * L_ * 2);
  float* S     = (float*)walloc((size_t)NSL_ * L_ * L_ * 4);
  bf16*  At    = (bf16*)walloc((size_t)NSL_ * L_ * L_ * 2);
  bf16*  attnm = (bf16*)walloc((size_t)BL_ * D_ * 2);
  float* proj  = (float*)walloc((size_t)BL_ * D_ * 4);
  float* out1F = (float*)walloc((size_t)BL_ * D_ * 4);
  bf16*  out1B = (bf16*)walloc((size_t)BL_ * D_ * 2);
  bf16*  hmid  = (bf16*)walloc((size_t)BL_ * DFF_ * 2);
  float* ffn2o = (float*)walloc((size_t)BL_ * D_ * 4);
  float* pmax  = (float*)walloc((size_t)NSL_ * 32 * 4);
  float* gmax  = (float*)walloc((size_t)NSL_ * 4);
  float* psum  = (float*)walloc((size_t)NSL_ * 256 * 4);
  float* gscale= (float*)walloc((size_t)NSL_ * 4);

  // input + weight conversions
  cvt_kernel<<<(BL_ * D_) / 256, 256, 0, stream>>>(pro_inp, inpro, BL_ * D_);
  cvt_kernel<<<(BL_ * D_) / 256, 256, 0, stream>>>(rna_inp, inrna, BL_ * D_);
  for (int s = 0; s < 2; s++) {
    int pb = PB[s];
    wconv_kernel<<<(D_ * D_) / 256, 256, 0, stream>>>((const float*)d_in[pb + 0], WqT[s], D_, D_);
    wconv_kernel<<<(D_ * D_) / 256, 256, 0, stream>>>((const float*)d_in[pb + 2], WkT[s], D_, D_);
    wconv_kernel<<<(D_ * D_) / 256, 256, 0, stream>>>((const float*)d_in[pb + 4], WvT[s], D_, D_);
    wconv_kernel<<<(D_ * D_) / 256, 256, 0, stream>>>((const float*)d_in[pb + 6], WoT[s], D_, D_);
    wconv_kernel<<<(D_ * DFF_) / 256, 256, 0, stream>>>((const float*)d_in[pb + 8], W1T[s], D_, DFF_);
    wconv_kernel<<<(DFF_ * D_) / 256, 256, 0, stream>>>((const float*)d_in[pb + 10], W2T[s], DFF_, D_);
  }

  auto launch_gemm = [&](const bf16* A, int lda, const bf16* Bt, int ldb,
                         const float* bias, float* Cf, bf16* Cb, int ldc,
                         int M, int N, int Kd, int relu) {
    dim3 g(N / 32, M / 64);
    gemm_kernel<<<g, 32, 0, stream>>>(A, lda, Bt, ldb, bias, Cf, Cb, ldc, Kd, relu);
  };

  auto run_side = [&](const bf16* qinB, const bf16* kvinB, const float* qinF,
                      int s, float* outF) {
    int pb = PB[s];
    const float* wqb = (const float*)d_in[pb + 1];
    const float* wkb = (const float*)d_in[pb + 3];
    const float* wvb = (const float*)d_in[pb + 5];
    const float* wob = (const float*)d_in[pb + 7];
    const float* f1b = (const float*)d_in[pb + 9];
    const float* f2b = (const float*)d_in[pb + 11];
    const float* l1g = (const float*)d_in[pb + 12];
    const float* l1b = (const float*)d_in[pb + 13];
    const float* l2g = (const float*)d_in[pb + 14];
    const float* l2b = (const float*)d_in[pb + 15];
    const float* ca  = (const float*)d_in[pb + 16];
    const float* cg  = (const float*)d_in[pb + 17];

    // projections
    launch_gemm(qinB,  D_, WqT[s], D_, wqb, nullptr, Qbf, D_, BL_, D_, D_, 0);
    launch_gemm(kvinB, D_, WkT[s], D_, wkb, nullptr, Kbf, D_, BL_, D_, D_, 0);
    gemm_vt_kernel<<<dim3(D_ / 32, BL_ / 64), 32, 0, stream>>>(kvinB, D_, WvT[s], D_, wvb, Vt, D_);
    // logits
    qk_kernel<<<dim3(L_ / 32, L_ / 64, NSL_), 32, 0, stream>>>(Qbf, Kbf, hb, pi, mask, ca, cg, S);
    // joint 2-D softmax (max, exp+transpose, sum -> scale = nz/sum)
    pmax_kernel<<<dim3(32, NSL_), 256, 0, stream>>>(S, pmax);
    fmax_kernel<<<NSL_, 32, 0, stream>>>(pmax, gmax);
    expT_kernel<<<dim3(L_ / 64, L_ / 64, NSL_), 256, 0, stream>>>(S, gmax, At, psum);
    fsum_kernel<<<NSL_, 256, 0, stream>>>(psum, protok, gscale);
    // out = scale * A^T V, merged heads
    av_kernel<<<dim3(DH_ / 32, L_ / 64, NSL_), 32, 0, stream>>>(At, Vt, gscale, attnm);
    // Wo + residual + LN1
    launch_gemm(attnm, D_, WoT[s], D_, wob, proj, nullptr, D_, BL_, D_, D_, 0);
    add_ln_kernel<<<BL_, 256, 0, stream>>>(qinF, proj, l1g, l1b, out1F, out1B);
    // FFN + residual + LN2
    launch_gemm(out1B, D_, W1T[s], D_, f1b, nullptr, hmid, DFF_, BL_, DFF_, D_, 1);
    launch_gemm(hmid, DFF_, W2T[s], DFF_, f2b, ffn2o, nullptr, D_, BL_, D_, DFF_, 0);
    add_ln_kernel<<<BL_, 256, 0, stream>>>(out1F, ffn2o, l2g, l2b, outF, nullptr);
  };

  float* outPro = (float*)d_out;
  float* outRna = (float*)d_out + (size_t)BL_ * D_;
  // pro_out: q = pro, kv = rna, params_pro (s=0)
  run_side(inpro, inrna, pro_inp, 0, outPro);
  // rna_out: q = rna, kv = pro, params_rna (s=1)
  run_side(inrna, inpro, rna_inp, 1, outRna);
}
